// AgentAttention_4260607557732
// MI455X (gfx1250) — matrix operand order repaired and checked by reading, not compiled, not run on hardware
//
#include <hip/hip_runtime.h>
#include <stdint.h>

typedef __attribute__((ext_vector_type(16))) __bf16 v16bf;
typedef __attribute__((ext_vector_type(8)))  float  v8f;

union BfFrag { v16bf v; uint32_t w[8]; };

__device__ __forceinline__ uint32_t bfbits(float f) {
  uint32_t x = __float_as_uint(f);
  return (x + 0x7FFFu + ((x >> 16) & 1u)) >> 16;   // RNE fp32 -> bf16
}
__device__ __forceinline__ uint32_t packbf(float lo, float hi) {
  return bfbits(lo) | (bfbits(hi) << 16);
}
__device__ __forceinline__ v8f wmma_bf16(const BfFrag& a, const BfFrag& b, v8f c) {
  return __builtin_amdgcn_wmma_f32_16x16x32_bf16(false, a.v, false, b.v, (short)0, c,
                                                 false, false);
}
// A fragment: 16x32 bf16.  lane m = l&15, half = l>>4.
// VGPR v in [0,4): K = 8*half + 2v ; v in [4,8): K = 16 + 8*half + 2(v-4)  (ISA 7.12.2)
__device__ __forceinline__ void frag_a(BfFrag& f, const uint32_t* base, int stride,
                                       int m0, int kp0, int lane) {
  const int m = m0 + (lane & 15);
  const int half = lane >> 4;
  const uint32_t* row = base + m * stride + kp0;
#pragma unroll
  for (int v = 0; v < 8; ++v) {
    const int kb = (v < 4) ? (8 * half + 2 * v) : (16 + 8 * half + 2 * (v - 4));
    f.w[v] = row[kb >> 1];
  }
}
// B fragment: 32x16 bf16.  lane n = l&15, half = l>>4.  VGPR v: same K as frag_a.
__device__ __forceinline__ void frag_b(BfFrag& f, const uint32_t* base, int stride,
                                       int n0, int kp0, int lane) {
  const int n = n0 + (lane & 15);
  const int half = lane >> 4;
  const uint32_t* row = base + n * stride + kp0;
#pragma unroll
  for (int v = 0; v < 8; ++v) f.w[v] = row[(v < 4) ? (4 * half + v) : (8 + 4 * half + (v - 4))];
}

// ---------------------------------------------------------------- constants
#define CB 64            // batch
#define CN 785           // tokens
#define CC_ 768          // channels
#define CH 12            // heads
#define CD 64            // head dim
#define CA 49            // agents
#define CW 28            // window
#define CBH (CB * CH)    // 768
#define CM (CB * CN)     // 50240 rows

// ============================================================ bias precompute
__device__ __forceinline__ float bilin7(const float* __restrict__ p, int i, int j) {
  const float fy = (i + 0.5f) * 0.25f - 0.5f;
  const float fx = (j + 0.5f) * 0.25f - 0.5f;
  int y0 = (int)floorf(fy), x0 = (int)floorf(fx);
  const float wy = fy - (float)y0, wx = fx - (float)x0;
  int ya = y0 < 0 ? 0 : (y0 > 6 ? 6 : y0);
  int yb = (y0 + 1) < 0 ? 0 : ((y0 + 1) > 6 ? 6 : y0 + 1);
  int xa = x0 < 0 ? 0 : (x0 > 6 ? 6 : x0);
  int xb = (x0 + 1) < 0 ? 0 : ((x0 + 1) > 6 ? 6 : x0 + 1);
  return (1.f - wy) * ((1.f - wx) * p[ya * 7 + xa] + wx * p[ya * 7 + xb]) +
         wy * ((1.f - wx) * p[yb * 7 + xa] + wx * p[yb * 7 + xb]);
}

__global__ void posbias_kernel(const float* __restrict__ an, const float* __restrict__ ah,
                               const float* __restrict__ aw, const float* __restrict__ ac,
                               float* __restrict__ pos_bias) {
  const int idx = blockIdx.x * 256 + threadIdx.x;      // (h, a, n)
  if (idx >= CH * CA * CN) return;
  const int n = idx % CN;
  const int ha = idx / CN;
  const int a = ha % CA, h = ha / CA;
  float v;
  if (n == 0) {
    v = ac[h * CA + a];
  } else {
    const int nn = n - 1, i = nn / CW, j = nn % CW;
    v = bilin7(an + (h * CA + a) * 49, i, j) + ah[(h * CA + a) * CW + i] +
        aw[(h * CA + a) * CW + j];
  }
  pos_bias[idx] = v;
}

__global__ void agentbias_kernel(const float* __restrict__ na, const float* __restrict__ hb,
                                 const float* __restrict__ wb, const float* __restrict__ ca,
                                 float* __restrict__ agent_bias) {
  const int idx = blockIdx.x * 256 + threadIdx.x;      // (h, n, a)
  if (idx >= CH * CN * CA) return;
  const int a = idx % CA;
  const int rest = idx / CA;
  const int n = rest % CN, h = rest / CN;
  float v;
  if (n == 0) {
    v = ca[h * CA + a];
  } else {
    const int nn = n - 1, i = nn / CW, j = nn % CW;
    v = bilin7(na + (h * CA + a) * 49, i, j) + hb[(h * CW + i) * CA + a] +
        wb[(h * CW + j) * CA + a];
  }
  agent_bias[idx] = v;
}

// ============================================================ generic GEMM A*B^T
// A: (M,K) fp32 row-major, Bm: (Nout,K) fp32 row-major; out (M,Nout); K%32==0, Nout%128==0.
// Register double-buffered: next K-tile's global loads issue before the compute of the
// current tile so vmem latency overlaps the 8 WMMAs; L2 warmed 2 tiles ahead via prefetch.
__global__ __launch_bounds__(256) void gemm_abT_kernel(
    const float* __restrict__ A, long lda, const float* __restrict__ Bm, long ldb,
    float* __restrict__ out, long ldo, int M, int K, const float* __restrict__ bias) {
  __shared__ uint32_t Au[128][17];
  __shared__ uint32_t Bu[128][17];
  const int t = threadIdx.x, lane = t & 31, wave = t >> 5;
  const long m0 = (long)blockIdx.y * 128;
  const long n0 = (long)blockIdx.x * 128;
  const int wm = (wave & 1) * 64, wn = (wave >> 1) * 32;
  const v8f vz = {0.f, 0.f, 0.f, 0.f, 0.f, 0.f, 0.f, 0.f};
  v8f acc[4][2];
#pragma unroll
  for (int i = 0; i < 4; ++i) { acc[i][0] = vz; acc[i][1] = vz; }
  const int lr = t >> 1, kh = (t & 1) * 16;
  const long gm = m0 + lr;
  const bool aval = gm < M;
  const float* pa = A + (aval ? gm * lda : 0) + kh;
  const float* pb = Bm + (n0 + lr) * ldb + kh;
  float fa[16], fb[16];
#pragma unroll
  for (int i = 0; i < 16; ++i) fa[i] = aval ? pa[i] : 0.f;
#pragma unroll
  for (int i = 0; i < 16; ++i) fb[i] = pb[i];
  for (int k0 = 0; k0 < K; k0 += 32) {
    __syncthreads();
#pragma unroll
    for (int c = 0; c < 8; ++c) Au[lr][(kh >> 1) + c] = packbf(fa[2 * c], fa[2 * c + 1]);
#pragma unroll
    for (int c = 0; c < 8; ++c) Bu[lr][(kh >> 1) + c] = packbf(fb[2 * c], fb[2 * c + 1]);
    __syncthreads();
    if (k0 + 32 < K) {  // issue next tile's loads; waited on only at next store
      const float* qa = pa + k0 + 32;
      const float* qb = pb + k0 + 32;
#pragma unroll
      for (int i = 0; i < 16; ++i) fa[i] = aval ? qa[i] : 0.f;
#pragma unroll
      for (int i = 0; i < 16; ++i) fb[i] = qb[i];
      if (k0 + 64 < K) {  // warm L2 two tiles ahead (global_prefetch_b8)
        __builtin_prefetch(pa + k0 + 64, 0, 3);
        __builtin_prefetch(pb + k0 + 64, 0, 3);
      }
    }
    BfFrag af[4], bf[2];
#pragma unroll
    for (int i = 0; i < 4; ++i) frag_a(af[i], &Au[0][0], 17, wm + 16 * i, 0, lane);
#pragma unroll
    for (int j = 0; j < 2; ++j) frag_b(bf[j], &Bu[0][0], 17, wn + 16 * j, 0, lane);
#pragma unroll
    for (int i = 0; i < 4; ++i)
#pragma unroll
      for (int j = 0; j < 2; ++j) acc[i][j] = wmma_bf16(af[i], bf[j], acc[i][j]);
  }
  const int ncol = lane & 15, rh = (lane >> 4) * 8;
#pragma unroll
  for (int i = 0; i < 4; ++i)
#pragma unroll
    for (int j = 0; j < 2; ++j) {
      const long col = n0 + wn + 16 * j + ncol;
      const float bv = bias ? bias[col] : 0.f;
#pragma unroll
      for (int r = 0; r < 8; ++r) {
        const long gmr = m0 + wm + 16 * i + rh + r;
        if (gmr < M) out[gmr * ldo + col] = acc[i][j][r] + bv;
      }
    }
}

// ============================================================ agent pooling (4x4 mean)
__global__ void pool_kernel(const float* __restrict__ qkv, float* __restrict__ agent) {
  const int idx = blockIdx.x * 256 + threadIdx.x;      // ((b*12+h)*49+a)*64+d
  if (idx >= CBH * CA * CD) return;
  const int d = idx & 63;
  const int r1 = idx >> 6;
  const int a = r1 % CA;
  const int bh = r1 / CA;
  const int b = bh / CH, h = bh % CH;
  const int py = a / 7, px = a % 7;
  const int c = h * CD + d;
  float s = 0.f;
#pragma unroll
  for (int iy = 0; iy < 4; ++iy)
#pragma unroll
    for (int ix = 0; ix < 4; ++ix) {
      const int tok = 1 + (py * 4 + iy) * CW + (px * 4 + ix);
      s += qkv[((long)(b * CN + tok)) * 2304 + c];
    }
  agent[idx] = s * (1.f / 16.f);
}

// ============================================================ stage 1: scores (49x785)
__global__ __launch_bounds__(256) void s1_scores_kernel(
    const float* __restrict__ qkv, const float* __restrict__ agent,
    const float* __restrict__ pos_bias, float* __restrict__ s1s) {
  __shared__ uint32_t Au[64][33];    // agent (scaled) 64x64, pairs along K
  __shared__ uint32_t Bu[128][33];   // K rows 128x64
  const int bh = blockIdx.y;
  const int b = bh / CH, h = bh % CH;
  const int n0 = blockIdx.x * 128;
  const int t = threadIdx.x, lane = t & 31, wave = t >> 5;
  const float scale = 0.125f;
  {  // A: agent
    const int a = t >> 2, c0 = (t & 3) * 16;
    float fv[16];
    if (a < CA) {
      const float* p = agent + ((long)(bh * CA + a)) * CD + c0;
#pragma unroll
      for (int i = 0; i < 16; ++i) fv[i] = p[i] * scale;
    } else {
#pragma unroll
      for (int i = 0; i < 16; ++i) fv[i] = 0.f;
    }
#pragma unroll
    for (int c = 0; c < 8; ++c) Au[a][(c0 >> 1) + c] = packbf(fv[2 * c], fv[2 * c + 1]);
  }
  {  // B: K rows
    const int r = t >> 1, c0 = (t & 1) * 32;
    const int n = n0 + r;
    float fv[32];
    if (n < CN) {
      const float* p = qkv + ((long)(b * CN + n)) * 2304 + CC_ + h * CD + c0;
#pragma unroll
      for (int i = 0; i < 32; ++i) fv[i] = p[i];
    } else {
#pragma unroll
      for (int i = 0; i < 32; ++i) fv[i] = 0.f;
    }
#pragma unroll
    for (int c = 0; c < 16; ++c) Bu[r][(c0 >> 1) + c] = packbf(fv[2 * c], fv[2 * c + 1]);
  }
  __syncthreads();
  const v8f vz = {0.f, 0.f, 0.f, 0.f, 0.f, 0.f, 0.f, 0.f};
  v8f acc[4] = {vz, vz, vz, vz};
#pragma unroll
  for (int ks = 0; ks < 2; ++ks) {
    BfFrag bf;
    frag_b(bf, &Bu[0][0], 33, wave * 16, ks * 16, lane);
#pragma unroll
    for (int i = 0; i < 4; ++i) {
      BfFrag af;
      frag_a(af, &Au[0][0], 33, 16 * i, ks * 16, lane);
      acc[i] = wmma_bf16(af, bf, acc[i]);
    }
  }
  const int ncol = lane & 15, rh = (lane >> 4) * 8;
#pragma unroll
  for (int i = 0; i < 4; ++i)
#pragma unroll
    for (int r = 0; r < 8; ++r) {
      const int a = 16 * i + rh + r;
      const int n = n0 + wave * 16 + ncol;
      if (a < CA && n < CN)
        s1s[((long)(bh * CA + a)) * CN + n] = acc[i][r] + pos_bias[(h * CA + a) * CN + n];
    }
}

// ============================================================ softmax over N (785) -> bf16 P1 (pad to 800)
__global__ __launch_bounds__(256) void softmax1_kernel(const float* __restrict__ s1s,
                                                       uint16_t* __restrict__ P1) {
  const long row = blockIdx.x;
  const float* src = s1s + row * CN;
  uint16_t* dst = P1 + row * 800;
  __shared__ float red[256];
  const int t = threadIdx.x;
  float m = -3.0e38f;
  for (int i = t; i < CN; i += 256) m = fmaxf(m, src[i]);
  red[t] = m;
  __syncthreads();
  for (int s = 128; s > 0; s >>= 1) {
    if (t < s) red[t] = fmaxf(red[t], red[t + s]);
    __syncthreads();
  }
  m = red[0];
  __syncthreads();
  float sum = 0.f;
  for (int i = t; i < CN; i += 256) sum += __expf(src[i] - m);
  red[t] = sum;
  __syncthreads();
  for (int s = 128; s > 0; s >>= 1) {
    if (t < s) red[t] += red[t + s];
    __syncthreads();
  }
  const float inv = 1.f / red[0];
  for (int i = t; i < 800; i += 256) {
    const float v = (i < CN) ? __expf(src[i] - m) * inv : 0.f;
    dst[i] = (uint16_t)bfbits(v);
  }
}

// ============================================================ agent_v = P1 (49x800) @ V (800x64)
// Register double-buffered across the 25 K-steps.
__global__ __launch_bounds__(256) void s1_av_kernel(const uint16_t* __restrict__ P1,
                                                    const float* __restrict__ qkv,
                                                    float* __restrict__ agent_v) {
  __shared__ uint32_t Au[64][17];
  __shared__ uint32_t Bu[64][17];
  const int bh = blockIdx.x;
  const int b = bh / CH, h = bh % CH;
  const int t = threadIdx.x, lane = t & 31, wave = t >> 5;
  const int mi = (wave & 3) * 16;
  const int cj = (wave >> 2) * 32;
  const v8f vz = {0.f, 0.f, 0.f, 0.f, 0.f, 0.f, 0.f, 0.f};
  v8f acc[2] = {vz, vz};
  uint32_t areg[8];
  float breg[16];
  auto load_tile = [&](int k0) {
    if (t < 128) {  // A: P1 rows (bf16 direct copy)
      const int a = t >> 1, kht = (t & 1) * 16;
      if (a < CA) {
        const uint32_t* p = (const uint32_t*)(P1 + ((long)(bh * CA + a)) * 800 + k0 + kht);
#pragma unroll
        for (int c = 0; c < 8; ++c) areg[c] = p[c];
      } else {
#pragma unroll
        for (int c = 0; c < 8; ++c) areg[c] = 0u;
      }
    } else {  // B: V gather, pairs along token dim
      const int s = t - 128;
      const int d = s & 63;
      const int kpb = (s >> 6) * 8;
#pragma unroll
      for (int c = 0; c < 8; ++c) {
        const int kp = kpb + c;
        const int n1 = k0 + 2 * kp, n2 = n1 + 1;
        breg[2 * c] =
            (n1 < CN) ? qkv[((long)(b * CN + n1)) * 2304 + 1536 + h * CD + d] : 0.f;
        breg[2 * c + 1] =
            (n2 < CN) ? qkv[((long)(b * CN + n2)) * 2304 + 1536 + h * CD + d] : 0.f;
      }
    }
  };
  load_tile(0);
  for (int k0 = 0; k0 < 800; k0 += 32) {
    __syncthreads();
    if (t < 128) {
      const int a = t >> 1, kht = (t & 1) * 16;
#pragma unroll
      for (int c = 0; c < 8; ++c) Au[a][(kht >> 1) + c] = areg[c];
    } else {
      const int s = t - 128;
      const int d = s & 63, kpb = (s >> 6) * 8;
#pragma unroll
      for (int c = 0; c < 8; ++c) Bu[d][kpb + c] = packbf(breg[2 * c], breg[2 * c + 1]);
    }
    __syncthreads();
    if (k0 + 32 < 800) load_tile(k0 + 32);
    BfFrag af, bf0, bf1;
    frag_a(af, &Au[0][0], 17, mi, 0, lane);
    frag_b(bf0, &Bu[0][0], 17, cj, 0, lane);
    frag_b(bf1, &Bu[0][0], 17, cj + 16, 0, lane);
    acc[0] = wmma_bf16(af, bf0, acc[0]);
    acc[1] = wmma_bf16(af, bf1, acc[1]);
  }
  const int ncol = lane & 15, rh = (lane >> 4) * 8;
#pragma unroll
  for (int j = 0; j < 2; ++j)
#pragma unroll
    for (int r = 0; r < 8; ++r)
      agent_v[((long)bh * 64 + mi + rh + r) * 64 + cj + 16 * j + ncol] = acc[j][r];
}

// ============================================================ stage 2 scores (785x49 -> padded 64)
__global__ __launch_bounds__(256) void s2_scores_kernel(
    const float* __restrict__ qkv, const float* __restrict__ agent,
    const float* __restrict__ agent_bias, float* __restrict__ s2s) {
  __shared__ uint32_t Au[128][33];   // Q tile (scaled)
  __shared__ uint32_t Bu[64][33];    // agents
  const int bh = blockIdx.y;
  const int b = bh / CH, h = bh % CH;
  const int n0 = blockIdx.x * 128;
  const int t = threadIdx.x, lane = t & 31, wave = t >> 5;
  const float scale = 0.125f;
  {  // A: Q
    const int r = t >> 1, c0 = (t & 1) * 32;
    const int n = n0 + r;
    float fv[32];
    if (n < CN) {
      const float* p = qkv + ((long)(b * CN + n)) * 2304 + h * CD + c0;
#pragma unroll
      for (int i = 0; i < 32; ++i) fv[i] = p[i] * scale;
    } else {
#pragma unroll
      for (int i = 0; i < 32; ++i) fv[i] = 0.f;
    }
#pragma unroll
    for (int c = 0; c < 16; ++c) Au[r][(c0 >> 1) + c] = packbf(fv[2 * c], fv[2 * c + 1]);
  }
  {  // B: agents
    const int a = t >> 2, c0 = (t & 3) * 16;
    float fv[16];
    if (a < CA) {
      const float* p = agent + ((long)(bh * CA + a)) * CD + c0;
#pragma unroll
      for (int i = 0; i < 16; ++i) fv[i] = p[i];
    } else {
#pragma unroll
      for (int i = 0; i < 16; ++i) fv[i] = 0.f;
    }
#pragma unroll
    for (int c = 0; c < 8; ++c) Bu[a][(c0 >> 1) + c] = packbf(fv[2 * c], fv[2 * c + 1]);
  }
  __syncthreads();
  const int rm = (wave & 3) * 32;
  const int cn = (wave >> 2) * 32;
  const v8f vz = {0.f, 0.f, 0.f, 0.f, 0.f, 0.f, 0.f, 0.f};
  v8f acc[2][2] = {{vz, vz}, {vz, vz}};
#pragma unroll
  for (int ks = 0; ks < 2; ++ks) {
    BfFrag af[2], bf[2];
#pragma unroll
    for (int i = 0; i < 2; ++i) frag_a(af[i], &Au[0][0], 33, rm + 16 * i, ks * 16, lane);
#pragma unroll
    for (int j = 0; j < 2; ++j) frag_b(bf[j], &Bu[0][0], 33, cn + 16 * j, ks * 16, lane);
#pragma unroll
    for (int i = 0; i < 2; ++i)
#pragma unroll
      for (int j = 0; j < 2; ++j) acc[i][j] = wmma_bf16(af[i], bf[j], acc[i][j]);
  }
  const int ncol = lane & 15, rh = (lane >> 4) * 8;
#pragma unroll
  for (int i = 0; i < 2; ++i)
#pragma unroll
    for (int j = 0; j < 2; ++j)
#pragma unroll
      for (int r = 0; r < 8; ++r) {
        const int n = n0 + rm + 16 * i + rh + r;
        const int a = cn + 16 * j + ncol;
        if (n < CN) {
          const float v = (a < CA) ? acc[i][j][r] + agent_bias[(h * CN + n) * CA + a]
                                   : -1.0e30f;
          s2s[((long)bh * CN + n) * 64 + a] = v;
        }
      }
}

// ============================================================ softmax over A (row len 64, one wave per row)
__global__ __launch_bounds__(256) void softmax2_kernel(const float* __restrict__ s2s,
                                                       uint16_t* __restrict__ P2) {
  const long row = (long)blockIdx.x * 8 + (threadIdx.x >> 5);
  const int lane = threadIdx.x & 31;
  if (row >= (long)CBH * CN) return;
  const float* src = s2s + row * 64;
  const float v0 = src[lane], v1 = src[lane + 32];
  float m = fmaxf(v0, v1);
#pragma unroll
  for (int s = 16; s > 0; s >>= 1) m = fmaxf(m, __shfl_xor(m, s, 32));
  const float e0 = __expf(v0 - m), e1 = __expf(v1 - m);
  float sum = e0 + e1;
#pragma unroll
  for (int s = 16; s > 0; s >>= 1) sum += __shfl_xor(sum, s, 32);
  const float inv = 1.f / sum;
  uint16_t* dst = P2 + row * 64;
  dst[lane] = (uint16_t)bfbits(e0 * inv);
  dst[lane + 32] = (uint16_t)bfbits(e1 * inv);
}

// ============================================================ out = P2 (785x64) @ agent_v (64x64)
__global__ __launch_bounds__(256) void s2_out_kernel(const uint16_t* __restrict__ P2,
                                                     const float* __restrict__ agent_v,
                                                     float* __restrict__ attn_out) {
  __shared__ uint32_t Au[128][33];
  __shared__ uint32_t Bu[64][33];
  const int bh = blockIdx.y;
  const int b = bh / CH, h = bh % CH;
  const int n0 = blockIdx.x * 128;
  const int t = threadIdx.x, lane = t & 31, wave = t >> 5;
  {  // A: P2 rows bf16 direct
    const int r = t >> 1, c0 = (t & 1) * 32;
    const int n = n0 + r;
    if (n < CN) {
      const uint32_t* p = (const uint32_t*)(P2 + ((long)bh * CN + n) * 64 + c0);
#pragma unroll
      for (int c = 0; c < 16; ++c) Au[r][(c0 >> 1) + c] = p[c];
    } else {
#pragma unroll
      for (int c = 0; c < 16; ++c) Au[r][(c0 >> 1) + c] = 0u;
    }
  }
  {  // B: agent_v transposed gather, pairs along agent dim
    const int d = t & 63;
    const int kpb = (t >> 6) * 8;
    const float* av = agent_v + (long)bh * 4096;
#pragma unroll
    for (int c = 0; c < 8; ++c) {
      const int kp = kpb + c;
      Bu[d][kp] = packbf(av[(2 * kp) * 64 + d], av[(2 * kp + 1) * 64 + d]);
    }
  }
  __syncthreads();
  const int rm = (wave & 3) * 32;
  const int cn = (wave >> 2) * 32;
  const v8f vz = {0.f, 0.f, 0.f, 0.f, 0.f, 0.f, 0.f, 0.f};
  v8f acc[2][2] = {{vz, vz}, {vz, vz}};
#pragma unroll
  for (int ks = 0; ks < 2; ++ks) {
    BfFrag af[2], bf[2];
#pragma unroll
    for (int i = 0; i < 2; ++i) frag_a(af[i], &Au[0][0], 33, rm + 16 * i, ks * 16, lane);
#pragma unroll
    for (int j = 0; j < 2; ++j) frag_b(bf[j], &Bu[0][0], 33, cn + 16 * j, ks * 16, lane);
#pragma unroll
    for (int i = 0; i < 2; ++i)
#pragma unroll
      for (int j = 0; j < 2; ++j) acc[i][j] = wmma_bf16(af[i], bf[j], acc[i][j]);
  }
  const int ncol = lane & 15, rh = (lane >> 4) * 8;
#pragma unroll
  for (int i = 0; i < 2; ++i)
#pragma unroll
    for (int j = 0; j < 2; ++j)
#pragma unroll
      for (int r = 0; r < 8; ++r) {
        const int n = n0 + rm + 16 * i + rh + r;
        if (n < CN)
          attn_out[((long)(b * CN + n)) * CC_ + h * CD + cn + 16 * j + ncol] = acc[i][j][r];
      }
}

// ============================================================ depthwise 3x3 conv add
__global__ void dwc_kernel(const float* __restrict__ qkv, const float* __restrict__ w,
                           const float* __restrict__ bias, float* __restrict__ attn_out) {
  const long idx = (long)blockIdx.x * 256 + threadIdx.x;   // (b, pix, c)
  if (idx >= (long)CB * 784 * CC_) return;
  const int c = (int)(idx % CC_);
  const long rest = idx / CC_;
  const int pix = (int)(rest % 784);
  const int b = (int)(rest / 784);
  const int y = pix / CW, x = pix % CW;
  float s = bias[c];
#pragma unroll
  for (int dy = -1; dy <= 1; ++dy)
#pragma unroll
    for (int dx = -1; dx <= 1; ++dx) {
      const int yy = y + dy, xx = x + dx;
      if (yy >= 0 && yy < CW && xx >= 0 && xx < CW)
        s += qkv[((long)(b * CN + 1 + yy * CW + xx)) * 2304 + 1536 + c] *
             w[((dy + 1) * 3 + (dx + 1)) * CC_ + c];
    }
  attn_out[((long)(b * CN + 1 + pix)) * CC_ + c] += s;
}

// ============================================================ host side
static inline size_t al256(size_t v) { return (v + 255) & ~(size_t)255; }

extern "C" void kernel_launch(void* const* d_in, const int* in_sizes, int n_in,
                              void* d_out, int out_size, void* d_ws, size_t ws_size,
                              hipStream_t stream) {
  (void)in_sizes; (void)n_in; (void)out_size; (void)ws_size;
  const float* x       = (const float*)d_in[0];
  const float* w_qkv   = (const float*)d_in[1];
  const float* w_proj  = (const float*)d_in[2];
  const float* b_proj  = (const float*)d_in[3];
  const float* dwc_w   = (const float*)d_in[4];
  const float* dwc_b   = (const float*)d_in[5];
  const float* an_bias = (const float*)d_in[6];
  const float* ah_bias = (const float*)d_in[7];
  const float* aw_bias = (const float*)d_in[8];
  const float* na_bias = (const float*)d_in[9];
  const float* ha_bias = (const float*)d_in[10];
  const float* wa_bias = (const float*)d_in[11];
  const float* ac_bias = (const float*)d_in[12];
  const float* ca_bias = (const float*)d_in[13];
  float* out = (float*)d_out;
  char* ws = (char*)d_ws;

  // workspace layout
  const size_t sz_qkv    = (size_t)CM * 2304 * 4;
  const size_t sz_attn   = (size_t)CM * CC_ * 4;
  const size_t sz_agent  = (size_t)CBH * CA * CD * 4;
  const size_t sz_agentv = (size_t)CBH * 64 * 64 * 4;
  const size_t sz_posb   = (size_t)CH * CA * CN * 4;
  const size_t sz_agb    = (size_t)CH * CN * CA * 4;
  const size_t sz_s1s    = (size_t)CBH * CA * CN * 4;
  const size_t sz_P1     = (size_t)CBH * CA * 800 * 2;
  const size_t sz_s2s    = (size_t)CBH * CN * 64 * 4;

  size_t off = 0;
  const size_t o_qkv = off;    off += al256(sz_qkv);
  const size_t o_attn = off;   off += al256(sz_attn);
  const size_t o_agent = off;  off += al256(sz_agent);
  const size_t o_agentv = off; off += al256(sz_agentv);
  const size_t o_posb = off;   off += al256(sz_posb);
  const size_t o_agb = off;    off += al256(sz_agb);
  const size_t o_region = off;                       // s1s+P1 union, later reused as s2s
  const size_t o_s1s = o_region;
  const size_t o_P1 = o_region + al256(sz_s1s);
  const size_t o_s2s = o_region;                     // alias (s1s/P1 dead by then)
  size_t region = al256(sz_s1s) + al256(sz_P1);
  if (region < al256(sz_s2s)) region = al256(sz_s2s);
  const size_t o_P2 = o_region + region;

  float* qkv      = (float*)(ws + o_qkv);
  float* attn     = (float*)(ws + o_attn);
  float* agent    = (float*)(ws + o_agent);
  float* agent_v  = (float*)(ws + o_agentv);
  float* pos_bias = (float*)(ws + o_posb);
  float* ag_bias  = (float*)(ws + o_agb);
  float* s1s      = (float*)(ws + o_s1s);
  uint16_t* P1    = (uint16_t*)(ws + o_P1);
  float* s2s      = (float*)(ws + o_s2s);
  uint16_t* P2    = (uint16_t*)(ws + o_P2);

  // 1) biases
  posbias_kernel<<<dim3((CH * CA * CN + 255) / 256), 256, 0, stream>>>(
      an_bias, ah_bias, aw_bias, ac_bias, pos_bias);
  agentbias_kernel<<<dim3((CH * CN * CA + 255) / 256), 256, 0, stream>>>(
      na_bias, ha_bias, wa_bias, ca_bias, ag_bias);
  // 2) QKV projection: (50240x768) @ (2304x768)^T
  gemm_abT_kernel<<<dim3(2304 / 128, (CM + 127) / 128), 256, 0, stream>>>(
      x, CC_, w_qkv, CC_, qkv, 2304, CM, CC_, nullptr);
  // 3) agent pooling
  pool_kernel<<<dim3((CBH * CA * CD) / 256), 256, 0, stream>>>(qkv, agent);
  // 4) stage 1
  s1_scores_kernel<<<dim3(7, CBH), 256, 0, stream>>>(qkv, agent, pos_bias, s1s);
  softmax1_kernel<<<dim3(CBH * CA), 256, 0, stream>>>(s1s, P1);
  s1_av_kernel<<<dim3(CBH), 256, 0, stream>>>(P1, qkv, agent_v);
  // 5) stage 2
  s2_scores_kernel<<<dim3(7, CBH), 256, 0, stream>>>(qkv, agent, ag_bias, s2s);
  softmax2_kernel<<<dim3((CBH * CN + 7) / 8), 256, 0, stream>>>(s2s, P2);
  s2_out_kernel<<<dim3(7, CBH), 256, 0, stream>>>(P2, agent_v, attn);
  // 6) depthwise conv add
  dwc_kernel<<<dim3(((long)CB * 784 * CC_) / 256), 256, 0, stream>>>(qkv, dwc_w, dwc_b, attn);
  // 7) output projection
  gemm_abT_kernel<<<dim3(CC_ / 128, (CM + 127) / 128), 256, 0, stream>>>(
      attn, CC_, w_proj, CC_, out, CC_, CM, CC_, b_proj);
}
